// MultiHeadAttention_6519760355389
// MI455X (gfx1250) — matrix operand order repaired and checked by reading, not compiled, not run on hardware
//
#include <hip/hip_runtime.h>

typedef __attribute__((ext_vector_type(16))) _Float16 v16h;
typedef __attribute__((ext_vector_type(8)))  _Float16 v8h;
typedef __attribute__((ext_vector_type(8)))  float    v8f;

static constexpr int kE  = 512;
static constexpr int kS  = 4096;
static constexpr int kB  = 2;
static constexpr int kH  = 8;
static constexpr int kD  = 64;
static constexpr int kBH = kB * kH;      // 16
static constexpr int kM  = kB * kS;      // 8192 rows of x
static constexpr int kLdp = 72;          // padded LDS row stride (halves)

// ---------------------------------------------------------------------------
// WMMA fragment helpers (CDNA5 16x16x32 f16 -> f32, ISA 7.12.2 layouts)
// ---------------------------------------------------------------------------

// A-matrix 16x32 f16: lane holds row (lane&15); K halves: kb..kb+7 and
// 16+kb..16+kb+7 with kb = 8*(lane>>4). Source: row-major, stride ld halves.
__device__ __forceinline__ v16h load_frag_a(const _Float16* p, int ld) {
  const int lane = threadIdx.x & 31;
  const int r    = lane & 15;
  const int kb   = (lane >> 4) << 3;
  const _Float16* row = p + (size_t)r * ld;
  v8h lo = *(const v8h*)(row + kb);
  v8h hi = *(const v8h*)(row + 16 + kb);
  v16h a;
#pragma unroll
  for (int i = 0; i < 8; ++i) { a[i] = lo[i]; a[i + 8] = hi[i]; }
  return a;
}

// B-matrix 32x16 f16: lane holds column (lane&15); K halves: kb..kb+7 and
// 16+kb..16+kb+7 with kb = 8*(lane>>4). Source is B^T row-major (N rows of K values), stride ld halves.
__device__ __forceinline__ v16h load_frag_b(const _Float16* p, int ld) {
  const int lane = threadIdx.x & 31;
  const int n    = lane & 15;
  const int ko   = (lane >> 4) << 3;
  const _Float16* row = p + (size_t)n * ld;
  v8h lo = *(const v8h*)(row + ko);
  v8h hi = *(const v8h*)(row + 16 + ko);
  v16h b;
#pragma unroll
  for (int i = 0; i < 8; ++i) { b[i] = lo[i]; b[i + 8] = hi[i]; }
  return b;
}

__device__ __forceinline__ v8f wmma_f16(v16h a, v16h b, v8f c) {
  return __builtin_amdgcn_wmma_f32_16x16x32_f16(false, a, false, b,
                                                (short)0, c, false, false);
}

// ---------------------------------------------------------------------------
// Stage 1: fp32 -> fp16 conversion
// ---------------------------------------------------------------------------
__global__ void f32_to_f16_kernel(const float* __restrict__ src,
                                  _Float16* __restrict__ dst, int n) {
  int i = blockIdx.x * blockDim.x + threadIdx.x;
  if (i < n) dst[i] = (_Float16)src[i];
}

// ---------------------------------------------------------------------------
// Stage 2: QKV projection.  out[m,n] = sum_k x[m,k]*W[n,k] + b[n]
// Wave tile: 32 rows x 64 cols (2 A-frags x 4 B-frags -> 8 WMMAs / k-step).
// mode 0: Q, scaled by 1/sqrt(D), layout [BH][S][D]
// mode 1: K, layout [BH][S][D]
// mode 2: V, layout transposed [BH][D][S]
// ---------------------------------------------------------------------------
__global__ __launch_bounds__(128)
void qkv_proj_kernel(const _Float16* __restrict__ xh,
                     const _Float16* __restrict__ wh,
                     const float* __restrict__ bias,
                     _Float16* __restrict__ out, int mode) {
  const int wave = threadIdx.x >> 5;
  const int m0 = blockIdx.x * 32;
  const int n0 = blockIdx.y * 256 + wave * 64;
  v8f acc[2][4] = {};
#pragma unroll 2
  for (int kk = 0; kk < kE; kk += 32) {
    const v16h a0 = load_frag_a(xh + (size_t)m0 * kE + kk, kE);
    const v16h a1 = load_frag_a(xh + (size_t)(m0 + 16) * kE + kk, kE);
#pragma unroll
    for (int t = 0; t < 4; ++t) {
      const v16h b = load_frag_b(wh + (size_t)(n0 + 16 * t) * kE + kk, kE);
      acc[0][t] = wmma_f16(a0, b, acc[0][t]);
      acc[1][t] = wmma_f16(a1, b, acc[1][t]);
    }
  }
  const int lane = threadIdx.x & 31;
  const int n  = lane & 15;
  const int rb = (lane >> 4) << 3;
#pragma unroll
  for (int t = 0; t < 4; ++t) {
    const int col = n0 + 16 * t + n;
    const int h = col >> 6;
    const int d = col & 63;
    const float bv = bias[col];
#pragma unroll
    for (int rblk = 0; rblk < 2; ++rblk) {
#pragma unroll
      for (int j = 0; j < 8; ++j) {
        const int m  = m0 + rblk * 16 + rb + j;
        const int b_ = m >> 12;
        const int s  = m & (kS - 1);
        float v = acc[rblk][t][j] + bv;
        if (mode == 0) v *= 0.125f;          // 1/sqrt(64)
        if (mode == 2)
          out[((size_t)(b_ * kH + h) * kD + d) * kS + s] = (_Float16)v;
        else
          out[((size_t)(b_ * kH + h) * kS + s) * kD + d] = (_Float16)v;
      }
    }
  }
}

// ---------------------------------------------------------------------------
// Stage 3: flash attention with cooperative double-buffered LDS staging.
// Block = 4 waves x 16 query rows; all waves share each 64-key K/V tile.
// Q prescaled; K [BH][S][D]; V transposed [BH][D][S]; out f16 [B*S][E].
// 16 WMMAs per softmax update (8 QK^T + 8 PV).
// ---------------------------------------------------------------------------
__global__ __launch_bounds__(128)
void attn_kernel(const _Float16* __restrict__ qh,
                 const _Float16* __restrict__ kh,
                 const _Float16* __restrict__ vt,
                 _Float16* __restrict__ attnh) {
  __shared__ __align__(16) _Float16 sK[2][64 * kLdp];  // keys x D
  __shared__ __align__(16) _Float16 sV[2][64 * kLdp];  // d-rows x keys
  __shared__ __align__(16) _Float16 pl[4][16 * kLdp];  // per-wave P staging
  const int tid  = threadIdx.x;
  const int wave = tid >> 5;
  const int lane = tid & 31;
  const int hi   = lane >> 4;
  const int cn   = lane & 15;
  const int bh = blockIdx.y;
  const int q0 = blockIdx.x * 64 + wave * 16;

  const _Float16* kbh = kh + (size_t)bh * kS * kD;
  const _Float16* vbh = vt + (size_t)bh * kD * kS;

  // cooperative staging coords: 2 threads per 64-half row, 32B each
  const int srow  = tid >> 1;          // 0..63
  const int shalf = (tid & 1) << 5;    // 0 or 32 halves

  const _Float16* qbase = qh + ((size_t)bh * kS + q0) * kD;
  const v16h qa0 = load_frag_a(qbase, kD);
  const v16h qa1 = load_frag_a(qbase + 32, kD);

  v8f o0 = {}, o1 = {}, o2 = {}, o3 = {};
  float mrow[8], lrow[8];
#pragma unroll
  for (int j = 0; j < 8; ++j) { mrow[j] = -1e30f; lrow[j] = 0.f; }
  _Float16* pw = &pl[wave][0];

  // preload tile 0 into LDS
  {
    v16h kreg = *(const v16h*)(kbh + (size_t)srow * kD + shalf);
    v16h vreg = *(const v16h*)(vbh + (size_t)srow * kS + shalf);
    *(v16h*)(&sK[0][srow * kLdp + shalf]) = kreg;
    *(v16h*)(&sV[0][srow * kLdp + shalf]) = vreg;
  }
  __syncthreads();

  for (int kt = 0; kt < kS; kt += 64) {
    const int  cur  = (kt >> 6) & 1;
    const bool more = (kt + 64) < kS;
    v16h kreg, vreg;
    if (more) {   // fetch next tile into registers while this tile computes
      kreg = *(const v16h*)(kbh + (size_t)(kt + 64 + srow) * kD + shalf);
      vreg = *(const v16h*)(vbh + (size_t)srow * kS + (kt + 64) + shalf);
      if (kt + 128 < kS) {             // global_prefetch_b8 for tile after next
        __builtin_prefetch(kbh + (size_t)(kt + 128 + srow) * kD + shalf, 0, 1);
        __builtin_prefetch(vbh + (size_t)srow * kS + (kt + 128) + shalf, 0, 1);
      }
    }
    const _Float16* skc = &sK[cur][0];
    const _Float16* svc = &sV[cur][0];

    v8f s0 = {}, s1 = {}, s2 = {}, s3 = {};
    s0 = wmma_f16(qa0, load_frag_b(skc,                 kLdp), s0);
    s0 = wmma_f16(qa1, load_frag_b(skc + 32,            kLdp), s0);
    s1 = wmma_f16(qa0, load_frag_b(skc + 16 * kLdp,      kLdp), s1);
    s1 = wmma_f16(qa1, load_frag_b(skc + 16 * kLdp + 32, kLdp), s1);
    s2 = wmma_f16(qa0, load_frag_b(skc + 32 * kLdp,      kLdp), s2);
    s2 = wmma_f16(qa1, load_frag_b(skc + 32 * kLdp + 32, kLdp), s2);
    s3 = wmma_f16(qa0, load_frag_b(skc + 48 * kLdp,      kLdp), s3);
    s3 = wmma_f16(qa1, load_frag_b(skc + 48 * kLdp + 32, kLdp), s3);

    // online softmax per row (16 columns of a row group live in 16 lanes)
#pragma unroll
    for (int j = 0; j < 8; ++j) {
      float t = fmaxf(fmaxf(s0[j], s1[j]), fmaxf(s2[j], s3[j]));
#pragma unroll
      for (int msk = 8; msk >= 1; msk >>= 1) t = fmaxf(t, __shfl_xor(t, msk, 32));
      const float mn = fmaxf(mrow[j], t);
      const float fs = __expf(mrow[j] - mn);
      const float p0 = __expf(s0[j] - mn);
      const float p1 = __expf(s1[j] - mn);
      const float p2 = __expf(s2[j] - mn);
      const float p3 = __expf(s3[j] - mn);
      float rs = (p0 + p1) + (p2 + p3);
#pragma unroll
      for (int msk = 8; msk >= 1; msk >>= 1) rs += __shfl_xor(rs, msk, 32);
      lrow[j] = lrow[j] * fs + rs;
      mrow[j] = mn;
      o0[j] *= fs; o1[j] *= fs; o2[j] *= fs; o3[j] *= fs;
      const int r = j + (hi << 3);
      pw[r * kLdp + cn]      = (_Float16)p0;   // C-layout -> LDS
      pw[r * kLdp + 16 + cn] = (_Float16)p1;
      pw[r * kLdp + 32 + cn] = (_Float16)p2;
      pw[r * kLdp + 48 + cn] = (_Float16)p3;
    }
    __builtin_amdgcn_wave_barrier();
    const v16h pa0 = load_frag_a(pw, kLdp);       // keys 0..31
    const v16h pa1 = load_frag_a(pw + 32, kLdp);  // keys 32..63
    o0 = wmma_f16(pa0, load_frag_b(svc,                 kLdp), o0);
    o0 = wmma_f16(pa1, load_frag_b(svc + 32,            kLdp), o0);
    o1 = wmma_f16(pa0, load_frag_b(svc + 16 * kLdp,      kLdp), o1);
    o1 = wmma_f16(pa1, load_frag_b(svc + 16 * kLdp + 32, kLdp), o1);
    o2 = wmma_f16(pa0, load_frag_b(svc + 32 * kLdp,      kLdp), o2);
    o2 = wmma_f16(pa1, load_frag_b(svc + 32 * kLdp + 32, kLdp), o2);
    o3 = wmma_f16(pa0, load_frag_b(svc + 48 * kLdp,      kLdp), o3);
    o3 = wmma_f16(pa1, load_frag_b(svc + 48 * kLdp + 32, kLdp), o3);

    if (more) {   // publish next tile to the alternate buffer
      *(v16h*)(&sK[1 - cur][srow * kLdp + shalf]) = kreg;
      *(v16h*)(&sV[1 - cur][srow * kLdp + shalf]) = vreg;
    }
    __syncthreads();
  }

  const int b_ = bh >> 3;
  const int hd = (bh & (kH - 1)) * kD;
#pragma unroll
  for (int j = 0; j < 8; ++j) {
    const float inv = 1.0f / lrow[j];
    const int r = q0 + j + (hi << 3);
    const size_t rowoff = (size_t)(b_ * kS + r) * kE + hd;
    attnh[rowoff + cn]      = (_Float16)(o0[j] * inv);
    attnh[rowoff + 16 + cn] = (_Float16)(o1[j] * inv);
    attnh[rowoff + 32 + cn] = (_Float16)(o2[j] * inv);
    attnh[rowoff + 48 + cn] = (_Float16)(o3[j] * inv);
  }
}

// ---------------------------------------------------------------------------
// Stage 4: output projection + bias + residual (fp32 out), 32x64 wave tile
// ---------------------------------------------------------------------------
__global__ __launch_bounds__(128)
void out_proj_kernel(const _Float16* __restrict__ ah,
                     const _Float16* __restrict__ woh,
                     const float* __restrict__ bo,
                     const float* __restrict__ x,
                     float* __restrict__ out) {
  const int wave = threadIdx.x >> 5;
  const int m0 = blockIdx.x * 32;
  const int n0 = blockIdx.y * 256 + wave * 64;
  v8f acc[2][4] = {};
#pragma unroll 2
  for (int kk = 0; kk < kE; kk += 32) {
    const v16h a0 = load_frag_a(ah + (size_t)m0 * kE + kk, kE);
    const v16h a1 = load_frag_a(ah + (size_t)(m0 + 16) * kE + kk, kE);
#pragma unroll
    for (int t = 0; t < 4; ++t) {
      const v16h b = load_frag_b(woh + (size_t)(n0 + 16 * t) * kE + kk, kE);
      acc[0][t] = wmma_f16(a0, b, acc[0][t]);
      acc[1][t] = wmma_f16(a1, b, acc[1][t]);
    }
  }
  const int lane = threadIdx.x & 31;
  const int n  = lane & 15;
  const int rb = (lane >> 4) << 3;
#pragma unroll
  for (int t = 0; t < 4; ++t) {
    const int col = n0 + 16 * t + n;
    const float bv = bo[col];
#pragma unroll
    for (int rblk = 0; rblk < 2; ++rblk) {
#pragma unroll
      for (int j = 0; j < 8; ++j) {
        const size_t idx = (size_t)(m0 + rblk * 16 + rb + j) * kE + col;
        out[idx] = acc[rblk][t][j] + bv + x[idx];
      }
    }
  }
}

// ---------------------------------------------------------------------------
extern "C" void kernel_launch(void* const* d_in, const int* in_sizes, int n_in,
                              void* d_out, int out_size, void* d_ws, size_t ws_size,
                              hipStream_t stream) {
  (void)in_sizes; (void)n_in; (void)out_size; (void)ws_size;
  const float* x  = (const float*)d_in[0];
  const float* Wq = (const float*)d_in[1];
  const float* bq = (const float*)d_in[2];
  const float* Wk = (const float*)d_in[3];
  const float* bk = (const float*)d_in[4];
  const float* Wv = (const float*)d_in[5];
  const float* bv = (const float*)d_in[6];
  const float* Wo = (const float*)d_in[7];
  const float* bo = (const float*)d_in[8];
  float* out = (float*)d_out;

  const size_t NX = (size_t)kM * kE;      // 4194304
  const size_t NW = (size_t)kE * kE;      // 262144
  _Float16* xh  = (_Float16*)d_ws;
  _Float16* wqh = xh  + NX;
  _Float16* wkh = wqh + NW;
  _Float16* wvh = wkh + NW;
  _Float16* woh = wvh + NW;
  _Float16* qh  = woh + NW;               // [BH][S][D]
  _Float16* khb = qh  + NX;               // [BH][S][D]
  _Float16* vth = khb + NX;               // [BH][D][S]
  _Float16* ah  = vth + NX;               // [B*S][E]

  f32_to_f16_kernel<<<(int)((NX + 255) / 256), 256, 0, stream>>>(x,  xh,  (int)NX);
  f32_to_f16_kernel<<<(int)((NW + 255) / 256), 256, 0, stream>>>(Wq, wqh, (int)NW);
  f32_to_f16_kernel<<<(int)((NW + 255) / 256), 256, 0, stream>>>(Wk, wkh, (int)NW);
  f32_to_f16_kernel<<<(int)((NW + 255) / 256), 256, 0, stream>>>(Wv, wvh, (int)NW);
  f32_to_f16_kernel<<<(int)((NW + 255) / 256), 256, 0, stream>>>(Wo, woh, (int)NW);

  dim3 gp(kM / 32, kE / 256);             // (256, 2)
  qkv_proj_kernel<<<gp, 128, 0, stream>>>(xh, wqh, bq, qh,  0);
  qkv_proj_kernel<<<gp, 128, 0, stream>>>(xh, wkh, bk, khb, 1);
  qkv_proj_kernel<<<gp, 128, 0, stream>>>(xh, wvh, bv, vth, 2);

  dim3 ga(kS / 64, kBH);                  // (64, 16)
  attn_kernel<<<ga, 128, 0, stream>>>(qh, khb, vth, ah);

  out_proj_kernel<<<gp, 128, 0, stream>>>(ah, woh, bo, x, out);
}
